// IAU_26431228739893
// MI455X (gfx1250) — matrix operand order repaired and checked by reading, not compiled, not run on hardware
//
#include <hip/hip_runtime.h>
#include <hip/hip_bf16.h>

// ---------------- problem constants ----------------
#define B_   4
#define C_   512
#define N_   256
#define C2_  256
#define K2_  512          // 2*C2
#define BCN_ (B_*C_*N_)   // 524288

typedef unsigned short u16;
typedef __attribute__((ext_vector_type(16))) __bf16 v16bf;
typedef __attribute__((ext_vector_type(8)))  float  v8f;

// ---------------- helpers ----------------
__device__ __forceinline__ u16 f2bf(float f) {
  unsigned int u = __builtin_bit_cast(unsigned int, f);
  u += 0x7FFFu + ((u >> 16) & 1u);           // round-to-nearest-even
  return (u16)(u >> 16);
}

__device__ __forceinline__ v16bf ldpair(const u16* p0, const u16* p1) {
  union { uint4 q[2]; v16bf v; } u;
  u.q[0] = *(const uint4*)p0;
  u.q[1] = *(const uint4*)p1;
  return u.v;
}

// C(16x16,f32) += A(16xK,bf16) * B(Kx16,bf16).
// Arow: this lane's A row (m = lane&15), contiguous in K.
// Brow: this lane's B column (n = lane&15) stored transposed, contiguous in K.
__device__ __forceinline__ void wave_gemm_bf16(const u16* __restrict__ Arow,
                                               const u16* __restrict__ Brow,
                                               int Ktot, v8f& acc) {
  const int lane = threadIdx.x & 31;
  const int kA = ((lane >> 4) & 1) * 16;   // A: lanes 16..31 hold K+16 group
  const int kB = ((lane >> 4) & 1) * 16;   // B: lanes 16..31 hold K+16 group
  for (int k = 0; k < Ktot; k += 32) {
    __builtin_prefetch(Arow + k + 256, 0, 3);   // WGP-scope global_prefetch_b8
    __builtin_prefetch(Brow + k + 256, 0, 3);
    v16bf a = ldpair(Arow + k + kA, Arow + k + kA + 8);
    v16bf b = ldpair(Brow + k + kB, Brow + k + kB + 8);
    acc = __builtin_amdgcn_wmma_f32_16x16x32_bf16(
        false, a, false, b, (short)0, acc, false, false);
  }
}

// 1x2 N-blocked variant: one A feed, two B tiles (Brow1 = Brow0 + 16 rows).
__device__ __forceinline__ void wave_gemm_bf16_x2(const u16* __restrict__ Arow,
                                                  const u16* __restrict__ Brow0,
                                                  const u16* __restrict__ Brow1,
                                                  int Ktot, v8f& acc0, v8f& acc1) {
  const int lane = threadIdx.x & 31;
  const int kA = ((lane >> 4) & 1) * 16;
  const int kB = ((lane >> 4) & 1) * 16;
  for (int k = 0; k < Ktot; k += 32) {
    __builtin_prefetch(Arow + k + 256, 0, 3);
    __builtin_prefetch(Brow0 + k + 256, 0, 3);
    __builtin_prefetch(Brow1 + k + 256, 0, 3);
    v16bf a  = ldpair(Arow + k + kA, Arow + k + kA + 8);
    v16bf b0 = ldpair(Brow0 + k + kB, Brow0 + k + kB + 8);
    v16bf b1 = ldpair(Brow1 + k + kB, Brow1 + k + kB + 8);
    acc0 = __builtin_amdgcn_wmma_f32_16x16x32_bf16(
        false, a, false, b0, (short)0, acc0, false, false);
    acc1 = __builtin_amdgcn_wmma_f32_16x16x32_bf16(
        false, a, false, b1, (short)0, acc1, false, false);
  }
}

// ---------------- conversion kernels ----------------
__global__ void k_cvt(const float* __restrict__ s, u16* __restrict__ d, int n) {
  int i = blockIdx.x * 256 + threadIdx.x;
  if (i < n) d[i] = f2bf(s[i]);
}
// src (Bn,R,Cc) f32 -> dst (Bn,Cc,R) bf16
__global__ void k_cvt_T(const float* __restrict__ s, u16* __restrict__ d,
                        int Bn, int R, int Cc) {
  int i = blockIdx.x * 256 + threadIdx.x;
  int tot = Bn * R * Cc;
  if (i >= tot) return;
  int c = i % Cc, r = (i / Cc) % R, b = i / (Cc * R);
  d[((size_t)b * Cc + c) * R + r] = f2bf(s[i]);
}

// ------- gram: G[b] = gx @ gx^T (C x C), K=N; 16x32 output per wave --------
__global__ void k_gram(const u16* __restrict__ GX, float* __restrict__ G) {
  int wave = (blockIdx.x * blockDim.x + threadIdx.x) >> 5;
  int lane = threadIdx.x & 31;
  int b = wave >> 9, t = wave & 511;            // 32 mt * 16 nt-pairs per batch
  int mt = t >> 4, ntp = t & 15;
  int n0 = ntp * 32;
  const u16* gx = GX + (size_t)b * C_ * N_;
  const u16* Arow  = gx + (size_t)(mt * 16 + (lane & 15)) * N_;
  const u16* Brow0 = gx + (size_t)(n0 + (lane & 15)) * N_;  // gx^T: k-contig
  const u16* Brow1 = Brow0 + (size_t)16 * N_;
  v8f a0 = {}, a1 = {};
  wave_gemm_bf16_x2(Arow, Brow0, Brow1, N_, a0, a1);
  int ncol = lane & 15, moff = (lane >> 4) * 8;
  float* Cb = G + (size_t)b * C_ * C_;
#pragma unroll
  for (int r = 0; r < 8; r++) {
    size_t rowoff = (size_t)(mt * 16 + moff + r) * C_;
    Cb[rowoff + n0 + ncol]      = a0[r];
    Cb[rowoff + n0 + 16 + ncol] = a1[r];
  }
}

// ---------------- row softmax over G (rows of length 512) -> bf16 F ----------
__global__ void k_softmax_g(const float* __restrict__ G, u16* __restrict__ F) {
  int row = blockIdx.x, t = threadIdx.x;
  const float* g = G + (size_t)row * 512;
  __shared__ float red[256];
  float v0 = g[t], v1 = g[t + 256];
  red[t] = fmaxf(v0, v1); __syncthreads();
  for (int s = 128; s > 0; s >>= 1) { if (t < s) red[t] = fmaxf(red[t], red[t + s]); __syncthreads(); }
  float mx = red[0]; __syncthreads();
  float e0 = __expf(v0 - mx), e1 = __expf(v1 - mx);
  red[t] = e0 + e1; __syncthreads();
  for (int s = 128; s > 0; s >>= 1) { if (t < s) red[t] += red[t + s]; __syncthreads(); }
  float inv = 1.0f / red[0];
  u16* o = F + (size_t)row * 512;
  o[t] = f2bf(e0 * inv); o[t + 256] = f2bf(e1 * inv);
}

// -------- Y^T = (F @ gx)^T stored bf16 (B,N,C); 16x32 output per wave -------
__global__ void k_y(const u16* __restrict__ F, const u16* __restrict__ GXT,
                    u16* __restrict__ YT) {
  int wave = (blockIdx.x * blockDim.x + threadIdx.x) >> 5;
  int lane = threadIdx.x & 31;
  int b = wave >> 8, t = wave & 255;            // 32 mt * 8 nt-pairs per batch
  int mt = t >> 3, ntp = t & 7;
  int n0 = ntp * 32;
  const u16* Arow  = F + (size_t)b * C_ * C_ + (size_t)(mt * 16 + (lane & 15)) * C_;
  const u16* Brow0 = GXT + (size_t)b * N_ * C_ + (size_t)(n0 + (lane & 15)) * C_;
  const u16* Brow1 = Brow0 + (size_t)16 * C_;
  v8f a0 = {}, a1 = {};
  wave_gemm_bf16_x2(Arow, Brow0, Brow1, C_, a0, a1);
  int ncol = lane & 15, moff = (lane >> 4) * 8;
  u16* o0 = YT + (size_t)b * N_ * C_ + (size_t)(n0 + ncol) * C_ + mt * 16 + moff;
  u16* o1 = o0 + (size_t)16 * C_;
#pragma unroll
  for (int r = 0; r < 8; r++) { o0[r] = f2bf(a0[r]); o1[r] = f2bf(a1[r]); }
}

// -------- T = w1 @ Y + w1_b  (B,C,N) f32; 16x32 output per wave -------------
__global__ void k_w1(const u16* __restrict__ W1, const u16* __restrict__ YT,
                     const float* __restrict__ w1_b, float* __restrict__ T) {
  int wave = (blockIdx.x * blockDim.x + threadIdx.x) >> 5;
  int lane = threadIdx.x & 31;
  int b = wave >> 8, t = wave & 255;
  int mt = t >> 3, ntp = t & 7;
  int n0 = ntp * 32;
  const u16* Arow  = W1 + (size_t)(mt * 16 + (lane & 15)) * C_;
  const u16* Brow0 = YT + (size_t)b * N_ * C_ + (size_t)(n0 + (lane & 15)) * C_;
  const u16* Brow1 = Brow0 + (size_t)16 * C_;
  v8f a0 = {}, a1 = {};
  wave_gemm_bf16_x2(Arow, Brow0, Brow1, C_, a0, a1);
  int ncol = lane & 15, moff = (lane >> 4) * 8;
#pragma unroll
  for (int r = 0; r < 8; r++) {
    int row = mt * 16 + moff + r;
    float bias = w1_b[row];
    size_t base = ((size_t)(b * C_ + row)) * N_;
    T[base + n0 + ncol]      = a0[r] + bias;
    T[base + n0 + 16 + ncol] = a1[r] + bias;
  }
}

// ---------------- per-channel BN stats: X(B,CH,256) -> mu/rs per channel -----
__global__ void k_stats(const float* __restrict__ X, int CH, int Bn,
                        float* __restrict__ mu, float* __restrict__ rs) {
  int ch = blockIdx.x, t = threadIdx.x;
  float s = 0.f, q = 0.f;
  for (int b = 0; b < Bn; b++) {
    float v = X[((size_t)(b * CH + ch)) * 256 + t];
    s += v; q += v * v;
  }
  __shared__ float r1[256], r2[256];
  r1[t] = s; r2[t] = q; __syncthreads();
  for (int k = 128; k > 0; k >>= 1) { if (t < k) { r1[t] += r1[t + k]; r2[t] += r2[t + k]; } __syncthreads(); }
  if (t == 0) {
    float cnt = (float)(Bn * 256);
    float m = r1[0] / cnt, var = r2[0] / cnt - m * m;
    mu[ch] = m; rs[ch] = rsqrtf(var + 1e-5f);
  }
}

// ---------------- z = x + bn(T); write Z f32 and Z^T bf16 ----------------
__global__ void k_z(const float* __restrict__ x, const float* __restrict__ T,
                    const float* __restrict__ mu, const float* __restrict__ rs,
                    const float* __restrict__ g, const float* __restrict__ be,
                    float* __restrict__ Z, u16* __restrict__ ZT) {
  int idx = blockIdx.x * 256 + threadIdx.x;     // B*C*N
  int n = idx & 255, c = (idx >> 8) & 511, b = idx >> 17;
  float tv = (T[idx] - mu[c]) * rs[c] * g[c] + be[c];
  float z = x[idx] + tv;
  Z[idx] = z;
  ZT[((size_t)(b * N_ + n)) * C_ + c] = f2bf(z);
}

// ---------------- u[b][c] = mean_n Z ----------------
__global__ void k_u(const float* __restrict__ Z, float* __restrict__ u) {
  int idx = blockIdx.x * 256 + threadIdx.x;     // B*C
  const float* zp = Z + (size_t)idx * N_;
  float s = 0.f;
  for (int n = 0; n < N_; n++) s += zp[n];
  u[idx] = s * (1.0f / N_);
}

// ---------------- a_raw = sa_w @ z + sa_b  (B,4,N) ----------------
__global__ void k_sa(const float* __restrict__ Z, const float* __restrict__ w,
                     const float* __restrict__ bb, float* __restrict__ A) {
  int idx = blockIdx.x * 256 + threadIdx.x;     // B*4*N
  int n = idx & 255, o = (idx >> 8) & 3, b = idx >> 10;
  float acc = bb[o];
  const float* zb = Z + (size_t)b * C_ * N_ + n;
  const float* wo = w + (size_t)o * C_;
  for (int c = 0; c < C_; c++) acc += wo[c] * zb[(size_t)c * N_];
  A[idx] = acc;
}

__global__ void k_aout(const float* __restrict__ A, const float* __restrict__ mu,
                       const float* __restrict__ rs, const float* __restrict__ g,
                       const float* __restrict__ be, float* __restrict__ out) {
  int idx = blockIdx.x * 256 + threadIdx.x;     // 4096
  int o = (idx >> 8) & 3;
  float v = (A[idx] - mu[o]) * rs[o] * g[o] + be[o];
  out[idx] = 1.0f / (1.0f + __expf(-v));
}

// ------------- xx = g_w @ Z + g_b : XXv f32 (B,C2,N), bf16 both layouts -----
__global__ void k_gproj(const u16* __restrict__ GW, const u16* __restrict__ ZT,
                        const float* __restrict__ g_b, float* __restrict__ XXv,
                        u16* __restrict__ XXvbf, u16* __restrict__ XRbf) {
  int wave = (blockIdx.x * blockDim.x + threadIdx.x) >> 5;
  int lane = threadIdx.x & 31;
  int b = wave >> 8, t = wave & 255;
  int mt = t >> 4, nt = t & 15;
  const u16* Arow = GW + (size_t)(mt * 16 + (lane & 15)) * C_;
  const u16* Brow = ZT + (size_t)b * N_ * C_ + (size_t)(nt * 16 + (lane & 15)) * C_;
  v8f acc = {};
  wave_gemm_bf16(Arow, Brow, C_, acc);
  int ncol = lane & 15, moff = (lane >> 4) * 8;
  int col = nt * 16 + ncol;
#pragma unroll
  for (int r = 0; r < 8; r++) {
    int row = mt * 16 + moff + r;
    float v = acc[r] + g_b[row];
    size_t i1 = ((size_t)(b * C2_ + row)) * N_ + col;
    XXv[i1] = v;
    XXvbf[i1] = f2bf(v);
    XRbf[((size_t)(b * N_ + col)) * C2_ + row] = f2bf(v);
  }
}

// ---------------- u2[b][o] = g_w[o]·u[b] + g_b[o] ----------------
__global__ void k_u2(const float* __restrict__ g_w, const float* __restrict__ g_b,
                     const float* __restrict__ u, float* __restrict__ u2) {
  int idx = blockIdx.x * 256 + threadIdx.x;     // B*C2
  int o = idx & 255, b = idx >> 8;
  float acc = g_b[o];
  const float* w = g_w + (size_t)o * C_;
  const float* ub = u + (size_t)b * C_;
  for (int c = 0; c < C_; c++) acc += w[c] * ub[c];
  u2[idx] = acc;
}

__global__ void k_gdot(const float* __restrict__ u2, const float* __restrict__ rel_w,
                       float* __restrict__ gdot) {
  int b = blockIdx.x, t = threadIdx.x;
  __shared__ float red[256];
  red[t] = u2[b * 256 + t] * rel_w[256 + t];
  __syncthreads();
  for (int s = 128; s > 0; s >>= 1) { if (t < s) red[t] += red[t + s]; __syncthreads(); }
  if (t == 0) gdot[b] = red[0];
}

// -------- score[b][i][j] = sum_c |xr[i,c]-xr[j,c]|*w_a[c] + gdot[b] + rel_b --
__global__ void k_score(const float* __restrict__ XXv, const float* __restrict__ rel_w,
                        const float* __restrict__ rel_b, const float* __restrict__ gdot,
                        float* __restrict__ SC) {
  int bi = blockIdx.x, b = bi >> 8, i = bi & 255, j = threadIdx.x;
  __shared__ float xri[256], wva[256];
  xri[j] = XXv[((size_t)(b * 256 + j)) * 256 + i];   // xr[b][i][j'] via (c,n) layout
  wva[j] = rel_w[j];
  __syncthreads();
  const float* col = XXv + (size_t)b * 65536 + j;
  float acc = 0.f;
  for (int c = 0; c < 256; c++) acc += fabsf(xri[c] - col[(size_t)c * 256]) * wva[c];
  SC[(size_t)bi * 256 + j] = acc + gdot[b] + rel_b[0];
}

// ---------------- global BN stats of score (two stage, deterministic) -------
__global__ void k_sc_part(const float* __restrict__ SC, float* __restrict__ ps,
                          float* __restrict__ pq) {
  int t = threadIdx.x, g0 = blockIdx.x * 256 + t;
  float s = 0.f, q = 0.f;
  for (int r = 0; r < 4; r++) { float v = SC[g0 + r * 65536]; s += v; q += v * v; }
  __shared__ float r1[256], r2[256];
  r1[t] = s; r2[t] = q; __syncthreads();
  for (int k = 128; k > 0; k >>= 1) { if (t < k) { r1[t] += r1[t + k]; r2[t] += r2[t + k]; } __syncthreads(); }
  if (t == 0) { ps[blockIdx.x] = r1[0]; pq[blockIdx.x] = r2[0]; }
}
__global__ void k_sc_fin(const float* __restrict__ ps, const float* __restrict__ pq,
                         float* __restrict__ mrs) {
  int t = threadIdx.x;
  __shared__ float r1[256], r2[256];
  r1[t] = ps[t]; r2[t] = pq[t]; __syncthreads();
  for (int k = 128; k > 0; k >>= 1) { if (t < k) { r1[t] += r1[t + k]; r2[t] += r2[t + k]; } __syncthreads(); }
  if (t == 0) {
    float cnt = 262144.f;
    float m = r1[0] / cnt, var = r2[0] / cnt - m * m;
    mrs[0] = m; mrs[1] = rsqrtf(var + 1e-5f);
  }
}

// -------- bn + diag mask + row softmax -> Wmat bf16 (B,N,N) ----------------
__global__ void k_score_softmax(const float* __restrict__ SC, const float* __restrict__ mrs,
                                const float* __restrict__ rel_g, const float* __restrict__ rel_be,
                                u16* __restrict__ Wbf) {
  int bi = blockIdx.x, i = bi & 255, j = threadIdx.x;
  float v = (SC[(size_t)bi * 256 + j] - mrs[0]) * mrs[1] * rel_g[0] + rel_be[0];
  if (j == i) v -= 1e8f;
  __shared__ float red[256];
  red[j] = v; __syncthreads();
  for (int s = 128; s > 0; s >>= 1) { if (j < s) red[j] = fmaxf(red[j], red[j + s]); __syncthreads(); }
  float mx = red[0]; __syncthreads();
  float e = __expf(v - mx);
  red[j] = e; __syncthreads();
  for (int s = 128; s > 0; s >>= 1) { if (j < s) red[j] += red[j + s]; __syncthreads(); }
  Wbf[(size_t)bi * 256 + j] = f2bf(e / red[0]);
}

// ---------------- xn = Wmat @ xr : bf16 (B,N,C2) ----------------
__global__ void k_xn(const u16* __restrict__ Wbf, const u16* __restrict__ XXvbf,
                     u16* __restrict__ XNbf) {
  int wave = (blockIdx.x * blockDim.x + threadIdx.x) >> 5;
  int lane = threadIdx.x & 31;
  int b = wave >> 8, t = wave & 255;
  int mt = t >> 4, nt = t & 15;
  const u16* Arow = Wbf + (size_t)b * N_ * N_ + (size_t)(mt * 16 + (lane & 15)) * N_;
  const u16* Brow = XXvbf + (size_t)b * C2_ * N_ + (size_t)(nt * 16 + (lane & 15)) * N_;
  v8f acc = {};
  wave_gemm_bf16(Arow, Brow, N_, acc);
  int ncol = lane & 15, moff = (lane >> 4) * 8;
  int col = nt * 16 + ncol;
#pragma unroll
  for (int r = 0; r < 8; r++) {
    int row = mt * 16 + moff + r;                       // i
    XNbf[((size_t)(b * N_ + row)) * C2_ + col] = f2bf(acc[r]);
  }
}

// ---------------- s = fsm_w @ [xr|xn]^T + fsm_b : (B,C2,N) f32 -------------
__global__ void k_fsm(const u16* __restrict__ FSM, const u16* __restrict__ XRbf,
                      const u16* __restrict__ XNbf, const float* __restrict__ fsm_b,
                      float* __restrict__ S) {
  int wave = (blockIdx.x * blockDim.x + threadIdx.x) >> 5;
  int lane = threadIdx.x & 31;
  int b = wave >> 8, t = wave & 255;
  int mt = t >> 4, nt = t & 15;
  const int kA = ((lane >> 4) & 1) * 16;
  const int kB = ((lane >> 4) & 1) * 16;
  const u16* Arow = FSM + (size_t)(mt * 16 + (lane & 15)) * K2_;
  const u16* xr = XRbf + ((size_t)(b * N_) + nt * 16 + (lane & 15)) * C2_;
  const u16* xn = XNbf + ((size_t)(b * N_) + nt * 16 + (lane & 15)) * C2_;
  v8f acc = {};
  for (int k = 0; k < K2_; k += 32) {
    const u16* br = (k < 256) ? (xr + k) : (xn + (k - 256));
    __builtin_prefetch(Arow + k + 256, 0, 3);
    v16bf a = ldpair(Arow + k + kA, Arow + k + kA + 8);
    v16bf bm = ldpair(br + kB, br + kB + 8);
    acc = __builtin_amdgcn_wmma_f32_16x16x32_bf16(
        false, a, false, bm, (short)0, acc, false, false);
  }
  int ncol = lane & 15, moff = (lane >> 4) * 8;
  int col = nt * 16 + ncol;
#pragma unroll
  for (int r = 0; r < 8; r++) {
    int row = mt * 16 + moff + r;
    S[((size_t)(b * C2_ + row)) * N_ + col] = acc[r] + fsm_b[row];
  }
}

// ---------------- yg[b][o] = mean_n relu(bn(s)) ----------------
__global__ void k_yg(const float* __restrict__ S, const float* __restrict__ mu,
                     const float* __restrict__ rs, const float* __restrict__ g,
                     const float* __restrict__ be, float* __restrict__ yg) {
  int idx = blockIdx.x * 256 + threadIdx.x;     // B*C2
  int o = idx & 255, b = idx >> 8;
  const float* sp = S + ((size_t)(b * C2_ + o)) * N_;
  float m = mu[o], r = rs[o], gg = g[o], bb = be[o];
  float s = 0.f;
  for (int n = 0; n < N_; n++) { float v = (sp[n] - m) * r * gg + bb; s += fmaxf(v, 0.f); }
  yg[idx] = s * (1.0f / N_);
}

// ---------------- y2 = ucat @ w2^T + w2_b ----------------
__global__ void k_y2(const float* __restrict__ w2, const float* __restrict__ w2_b,
                     const float* __restrict__ yg, const float* __restrict__ u2,
                     float* __restrict__ y2) {
  int idx = blockIdx.x * 256 + threadIdx.x;     // B*C
  int o = idx & 511, b = idx >> 9;
  float acc = w2_b[o];
  const float* w = w2 + (size_t)o * C_;
  const float* ygb = yg + b * 256;
  const float* u2b = u2 + b * 256;
  for (int c = 0; c < 256; c++) acc += w[c] * ygb[c];
  for (int c = 0; c < 256; c++) acc += w[256 + c] * u2b[c];
  y2[b * C_ + o] = acc;
}

// ---------------- BN over batch (4 samples) + affine ----------------
__global__ void k_y2bn(const float* __restrict__ y2, const float* __restrict__ g,
                       const float* __restrict__ be, float* __restrict__ y2n) {
  int o = blockIdx.x * 256 + threadIdx.x;       // 512
  float v[4]; float m = 0.f;
  for (int b = 0; b < 4; b++) { v[b] = y2[b * 512 + o]; m += v[b]; }
  m *= 0.25f;
  float var = 0.f;
  for (int b = 0; b < 4; b++) { float d = v[b] - m; var += d * d; }
  var *= 0.25f;
  float r = rsqrtf(var + 1e-5f) * g[o];
  for (int b = 0; b < 4; b++) y2n[b * 512 + o] = (v[b] - m) * r + be[o];
}

// ---------------- z2 = z + y2n broadcast ----------------
__global__ void k_final(const float* __restrict__ Z, const float* __restrict__ y2n,
                        float* __restrict__ out) {
  int idx = blockIdx.x * 256 + threadIdx.x;     // B*C*N
  int c = (idx >> 8) & 511, b = idx >> 17;
  out[idx] = Z[idx] + y2n[b * 512 + c];
}

// ============================================================================
extern "C" void kernel_launch(void* const* d_in, const int* in_sizes, int n_in,
                              void* d_out, int out_size, void* d_ws, size_t ws_size,
                              hipStream_t stream) {
  (void)in_sizes; (void)n_in; (void)out_size; (void)ws_size;
  const float* x      = (const float*)d_in[0];
  const float* g_w    = (const float*)d_in[1];
  const float* g_b    = (const float*)d_in[2];
  const float* w1_w   = (const float*)d_in[3];
  const float* w1_b   = (const float*)d_in[4];
  const float* w1_g   = (const float*)d_in[5];
  const float* w1_be  = (const float*)d_in[6];
  const float* rel_w  = (const float*)d_in[7];
  const float* rel_b  = (const float*)d_in[8];
  const float* rel_g  = (const float*)d_in[9];
  const float* rel_be = (const float*)d_in[10];
  const float* fsm_w  = (const float*)d_in[11];
  const float* fsm_b  = (const float*)d_in[12];
  const float* fsm_g  = (const float*)d_in[13];
  const float* fsm_be = (const float*)d_in[14];
  const float* w2_w   = (const float*)d_in[15];
  const float* w2_b   = (const float*)d_in[16];
  const float* w2_g   = (const float*)d_in[17];
  const float* w2_be  = (const float*)d_in[18];
  const float* sa_w   = (const float*)d_in[19];
  const float* sa_b   = (const float*)d_in[20];
  const float* sa_g   = (const float*)d_in[21];
  const float* sa_be  = (const float*)d_in[22];

  float* out = (float*)d_out;

  char* base = (char*)d_ws;
  size_t off = 0;
  auto alloc = [&](size_t bytes) -> void* {
    void* r = base + off;
    off = (off + bytes + 255) & ~(size_t)255;
    return r;
  };

  u16*   GXbf  = (u16*)  alloc((size_t)BCN_ * 2);
  u16*   GXTbf = (u16*)  alloc((size_t)BCN_ * 2);
  u16*   W1bf  = (u16*)  alloc((size_t)C_ * C_ * 2);
  u16*   GWbf  = (u16*)  alloc((size_t)C2_ * C_ * 2);
  u16*   FSMbf = (u16*)  alloc((size_t)C2_ * K2_ * 2);
  float* G     = (float*)alloc((size_t)B_ * C_ * C_ * 4);
  u16*   Fbf   = (u16*)  alloc((size_t)B_ * C_ * C_ * 2);
  u16*   YTbf  = (u16*)  alloc((size_t)BCN_ * 2);
  float* T     = (float*)alloc((size_t)BCN_ * 4);
  float* muT   = (float*)alloc(C_ * 4);
  float* rsT   = (float*)alloc(C_ * 4);
  float* Z     = (float*)alloc((size_t)BCN_ * 4);
  u16*   ZTbf  = (u16*)  alloc((size_t)BCN_ * 2);
  float* u     = (float*)alloc(B_ * C_ * 4);
  float* Araw  = (float*)alloc(B_ * 4 * N_ * 4);
  float* muA   = (float*)alloc(4 * 4);
  float* rsA   = (float*)alloc(4 * 4);
  float* XXv   = (float*)alloc((size_t)B_ * C2_ * N_ * 4);
  u16*   XXvbf = (u16*)  alloc((size_t)B_ * C2_ * N_ * 2);
  u16*   XRbf  = (u16*)  alloc((size_t)B_ * N_ * C2_ * 2);
  float* u2    = (float*)alloc(B_ * C2_ * 4);
  float* gdot  = (float*)alloc(B_ * 4);
  float* SC    = (float*)alloc((size_t)B_ * N_ * N_ * 4);
  float* ps    = (float*)alloc(256 * 4);
  float* pq    = (float*)alloc(256 * 4);
  float* mrs   = (float*)alloc(2 * 4);
  u16*   Wbf   = (u16*)  alloc((size_t)B_ * N_ * N_ * 2);
  u16*   XNbf  = (u16*)  alloc((size_t)B_ * N_ * C2_ * 2);
  float* S     = (float*)alloc((size_t)B_ * C2_ * N_ * 4);
  float* muS   = (float*)alloc(C2_ * 4);
  float* rsS   = (float*)alloc(C2_ * 4);
  float* yg    = (float*)alloc(B_ * C2_ * 4);
  float* y2    = (float*)alloc(B_ * C_ * 4);
  float* y2n   = (float*)alloc(B_ * C_ * 4);

  // bf16 operand prep
  k_cvt  <<<BCN_ / 256, 256, 0, stream>>>(x, GXbf, BCN_);
  k_cvt_T<<<BCN_ / 256, 256, 0, stream>>>(x, GXTbf, B_, C_, N_);
  k_cvt  <<<(C_ * C_) / 256, 256, 0, stream>>>(w1_w, W1bf, C_ * C_);
  k_cvt  <<<(C2_ * C_) / 256, 256, 0, stream>>>(g_w, GWbf, C2_ * C_);
  k_cvt  <<<(C2_ * K2_) / 256, 256, 0, stream>>>(fsm_w, FSMbf, C2_ * K2_);

  // attention over channels
  k_gram<<<256, 256, 0, stream>>>(GXbf, G);                        // 2048 waves x2 tiles
  k_softmax_g<<<B_ * C_, 256, 0, stream>>>(G, Fbf);
  k_y<<<128, 256, 0, stream>>>(Fbf, GXTbf, YTbf);                  // 1024 waves x2 tiles
  k_w1<<<128, 256, 0, stream>>>(W1bf, YTbf, w1_b, T);              // 1024 waves x2 tiles
  k_stats<<<C_, 256, 0, stream>>>(T, C_, B_, muT, rsT);
  k_z<<<BCN_ / 256, 256, 0, stream>>>(x, T, muT, rsT, w1_g, w1_be, Z, ZTbf);

  // sigmoid attention map (output #2)
  k_sa<<<(B_ * 4 * N_) / 256, 256, 0, stream>>>(Z, sa_w, sa_b, Araw);
  k_stats<<<4, 256, 0, stream>>>(Araw, 4, B_, muA, rsA);
  k_aout<<<(B_ * 4 * N_) / 256, 256, 0, stream>>>(Araw, muA, rsA, sa_g, sa_be,
                                                  out + (size_t)BCN_);

  // graph branch
  k_u<<<(B_ * C_) / 256, 256, 0, stream>>>(Z, u);
  k_gproj<<<128, 256, 0, stream>>>(GWbf, ZTbf, g_b, XXv, XXvbf, XRbf); // 1024 tiles
  k_u2<<<(B_ * C2_) / 256, 256, 0, stream>>>(g_w, g_b, u, u2);
  k_gdot<<<B_, 256, 0, stream>>>(u2, rel_w, gdot);
  k_score<<<B_ * N_, 256, 0, stream>>>(XXv, rel_w, rel_b, gdot, SC);
  k_sc_part<<<256, 256, 0, stream>>>(SC, ps, pq);
  k_sc_fin<<<1, 256, 0, stream>>>(ps, pq, mrs);
  k_score_softmax<<<B_ * N_, 256, 0, stream>>>(SC, mrs, rel_g, rel_be, Wbf);
  k_xn<<<128, 256, 0, stream>>>(Wbf, XXvbf, XNbf);                 // 1024 tiles
  k_fsm<<<128, 256, 0, stream>>>(FSMbf, XRbf, XNbf, fsm_b, S);     // 1024 tiles
  k_stats<<<C2_, 256, 0, stream>>>(S, C2_, B_, muS, rsS);
  k_yg<<<(B_ * C2_) / 256, 256, 0, stream>>>(S, muS, rsS, fsm_g, fsm_be, yg);
  k_y2<<<(B_ * C_) / 256, 256, 0, stream>>>(w2_w, w2_b, yg, u2, y2);
  k_y2bn<<<2, 256, 0, stream>>>(y2, w2_g, w2_be, y2n);
  k_final<<<BCN_ / 256, 256, 0, stream>>>(Z, y2n, out);
}
